// LSTM_78847009619979
// MI455X (gfx1250) — matrix operand order repaired and checked by reading, not compiled, not run on hardware
//
#include <hip/hip_runtime.h>
#include <hip/hip_bf16.h>

// ---------------------------------------------------------------------------
// Fused 3-layer LSTM (H=32, T=30) + affine head for MI455X (gfx1250, wave32).
//
//  * Gate GEMMs (x@W_ih^T and h@W_hh^T, K=32) -> v_wmma_f32_16x16x32_f16,
//    fp32 accumulate, 16 WMMAs per timestep per wave; C operand = inline 0.
//  * Bias folded into the activation FMA (pre-scaled by +-log2e) -> zero
//    extra instructions and zero C-register pressure.
//  * Activations on native trans ops (v_exp_f32/v_rcp_f32[/v_tanh_f32]),
//    branch-free, co-executing under the WMMA pipe.
//  * Inter-layer h-sequence lives in per-wave LDS (f16, in-place per layer).
//  * Same-wave LDS store->load ordering via s_wait_dscnt; no barriers.
//  * Head collapsed offline: Wf = W2@W1, bf = b2 + W2@b1 (affine of affine).
// ---------------------------------------------------------------------------

typedef __attribute__((ext_vector_type(16))) _Float16 v16h;
typedef __attribute__((ext_vector_type(8)))  _Float16 v8h;
typedef __attribute__((ext_vector_type(8)))  float    v8f;

#define HDIM  32
#define TSTEP 30
#define NGATE 128           // 4*H
#define NTILE 8             // 128/16 output tiles
#define ROWS_PER_WAVE 16
#define WAVES_PER_WG  2
#define NCLS  7

// packed layout per layer: f16 Wih[128][32] | f16 Whh[128][32] | f32 bias[128]
#define WMAT_HALFS (NGATE * HDIM)
#define LAYER_BYTES (WMAT_HALFS * 2 /*Wih*/ * 2 + NGATE * 4)
// after the 3 layer blocks: f32 Wf[7][32] | f32 bf[7]
#define HEAD_OFF (3 * LAYER_BYTES)

#define LOG2E 1.4426950408889634f

// ---------------------------------------------------------------------------
// Pack fp32 weights -> f16 in WMMA B-matrix lane order (blocks 0..2), and
// collapse the two dense head layers into one affine map (block 3).
// B tile j (32x16): lane L<16 holds K=0..15 of gate row (16j+L);
// lane L>=16 holds K=16..31 of gate row (16j+L-16). Row-major [gate][K]
// storage makes each lane's 16 halves contiguous -> one 32B load per tile.
// Layer 0 input dim (3) is zero-padded to K=32.
// ---------------------------------------------------------------------------
__global__ void pack_weights(const float* Wih0, const float* Whh0,
                             const float* bi0,  const float* bh0,
                             const float* Wih1, const float* Whh1,
                             const float* bi1,  const float* bh1,
                             const float* Wih2, const float* Whh2,
                             const float* bi2,  const float* bh2,
                             const float* W1,   const float* b1,
                             const float* W2,   const float* b2,
                             char* ws)
{
    const int l = blockIdx.x;
    if (l < 3) {
        const float* Wih = (l == 0) ? Wih0 : (l == 1) ? Wih1 : Wih2;
        const float* Whh = (l == 0) ? Whh0 : (l == 1) ? Whh1 : Whh2;
        const float* bi  = (l == 0) ? bi0  : (l == 1) ? bi1  : bi2;
        const float* bh  = (l == 0) ? bh0  : (l == 1) ? bh1  : bh2;
        const int din = (l == 0) ? 3 : HDIM;

        char* base = ws + (size_t)l * LAYER_BYTES;
        _Float16* wih_p  = (_Float16*)base;
        _Float16* whh_p  = (_Float16*)(base + WMAT_HALFS * 2);
        float*    bias_p = (float*)   (base + WMAT_HALFS * 4);

        for (int g = threadIdx.x; g < NGATE; g += blockDim.x) {
            for (int k = 0; k < HDIM; ++k) {
                wih_p[g * HDIM + k] = (k < din) ? (_Float16)Wih[g * din + k]
                                                : (_Float16)0.0f;
                whh_p[g * HDIM + k] = (_Float16)Whh[g * HDIM + k];
            }
            bias_p[g] = bi[g] + bh[g];
        }
    } else {
        // fused head: Wf = W2 @ W1 [7,32], bf = b2 + W2 @ b1 [7]
        float* Wf = (float*)(ws + HEAD_OFF);
        float* bf = Wf + NCLS * HDIM;
        for (int j = threadIdx.x; j < NCLS; j += blockDim.x) {
            for (int n = 0; n < HDIM; ++n) {
                float s = 0.0f;
                for (int k = 0; k < 16; ++k) s += W2[j * 16 + k] * W1[k * HDIM + n];
                Wf[j * HDIM + n] = s;
            }
            float s = b2[j];
            for (int k = 0; k < 16; ++k) s += W2[j * 16 + k] * b1[k];
            bf[j] = s;
        }
    }
}

// ---- branch-free activations with bias folded into the FMA ----------------
__device__ __forceinline__ float sigf_b(float x, float nb) {
    // sigmoid(x+b) = 1/(1 + 2^(-log2e*x + nb)),  nb = -log2e*b
    return __builtin_amdgcn_rcpf(
        1.0f + __builtin_amdgcn_exp2f(__builtin_fmaf(-LOG2E, x, nb)));
}
__device__ __forceinline__ float tanh_b(float x, float pb) {
    // tanh(x+b) = 1 - 2/(2^(2log2e*x + pb) + 1),  pb = 2log2e*b
    float e = __builtin_amdgcn_exp2f(__builtin_fmaf(2.0f * LOG2E, x, pb));
    return 1.0f - 2.0f * __builtin_amdgcn_rcpf(e + 1.0f);
}
__device__ __forceinline__ float tanh_c(float x) {
#if __has_builtin(__builtin_amdgcn_tanhf)
    return __builtin_amdgcn_tanhf(x);          // native v_tanh_f32
#else
    float e = __builtin_amdgcn_exp2f(2.0f * LOG2E * x);
    return 1.0f - 2.0f * __builtin_amdgcn_rcpf(e + 1.0f);
#endif
}

// ---------------------------------------------------------------------------
// Main fused kernel. blockDim = 64 (2 waves), each wave = 16 batch rows.
// ---------------------------------------------------------------------------
__global__ __launch_bounds__(WAVES_PER_WG * 32, 1)
void lstm3_wmma(const float* __restrict__ X, const char* __restrict__ wpack,
                float* __restrict__ out)
{
    // per-wave private h-sequence buffer: [T][16 rows][H] f16 (in-place per layer)
    __shared__ __align__(32) _Float16 seq_s[WAVES_PER_WG][TSTEP * ROWS_PER_WAVE * HDIM];

    const int lane   = threadIdx.x & 31;
    const int wave   = threadIdx.x >> 5;
    const int row0   = (blockIdx.x * WAVES_PER_WG + wave) * ROWS_PER_WAVE;
    const int mrow   = lane & 15;        // N for B/D, M for A, per ISA layouts
    const int hiHalf = lane >> 4;        // lane group 0/1
    _Float16* sbuf = seq_s[wave];

    for (int l = 0; l < 3; ++l) {
        const char* lb = wpack + (size_t)l * LAYER_BYTES;
        const _Float16* wih  = (const _Float16*)lb;
        const _Float16* whh  = (const _Float16*)(lb + WMAT_HALFS * 2);
        const float*    bias = (const float*)   (lb + WMAT_HALFS * 4);

        // Resident B tiles (16 x v16h = 128 VGPRs of f16) + pre-scaled biases
        v16h bih[NTILE], bhh[NTILE];
        float bsc[NTILE];
#pragma unroll
        for (int j = 0; j < NTILE; ++j) {
            const _Float16* p0 = wih + (16 * j + mrow) * HDIM + hiHalf * 16;
            const _Float16* p1 = whh + (16 * j + mrow) * HDIM + hiHalf * 16;
            bih[j] = *(const v16h*)p0;           // 32B per lane, L2 resident
            bhh[j] = *(const v16h*)p1;
            const float bv = bias[16 * j + mrow];
            // tiles 4,5 feed tanh (pre-scale +2log2e), others sigmoid (-log2e)
            bsc[j] = (j == 4 || j == 5) ? (2.0f * LOG2E) * bv : (-LOG2E) * bv;
        }

        float c1[8], c2[8];
#pragma unroll
        for (int r = 0; r < 8; ++r) { c1[r] = 0.0f; c2[r] = 0.0f; }
        v16h ah = {};                            // h_{t-1} in A layout (zero at t=0)

        for (int t = 0; t < TSTEP; ++t) {
            // ---- build A for the input GEMM -------------------------------
            v16h ain = {};
            if (l == 0) {
                // K padded 3 -> 32; only lanes 0..15 carry K=0..2
                if (lane < 16) {
                    const float* xp = X + ((size_t)(row0 + mrow) * TSTEP + t) * 3;
                    ain[0] = (_Float16)xp[0];
                    ain[1] = (_Float16)xp[1];
                    ain[2] = (_Float16)xp[2];
                }
            } else {
                // A layout: lane<16 halves = K0..15 ; lane>=16 = K16..31
                const _Float16* pa = sbuf + (t * ROWS_PER_WAVE + mrow) * HDIM + hiHalf * 16;
                v8h lo = *(const v8h*)pa;
                v8h hi = *(const v8h*)(pa + 8);
                ain = __builtin_shufflevector(lo, hi, 0,1,2,3,4,5,6,7,
                                                      8,9,10,11,12,13,14,15);
            }

            // ---- 16 WMMAs: gates = x@Wih^T + h@Whh^T (fp32 accum, C=0) ----
            v8f accv[NTILE];
            v8f zc = {};
#pragma unroll
            for (int j = 0; j < NTILE; ++j) {
                v8f d = __builtin_amdgcn_wmma_f32_16x16x32_f16(
                            false, ain, false, bih[j], (short)0, zc, false, false);
                accv[j] = __builtin_amdgcn_wmma_f32_16x16x32_f16(
                            false, ah,  false, bhh[j], (short)0, d,  false, false);
            }

            // ---- element-wise LSTM cell on the D layout -------------------
            // lane owns (M = r + 8*hiHalf, n = mrow and mrow+16), r = 0..7
            _Float16* hout = sbuf + (t * ROWS_PER_WAVE) * HDIM;
#pragma unroll
            for (int r = 0; r < 8; ++r) {
                const int M = r + 8 * hiHalf;
                float i1 = sigf_b(accv[0][r], bsc[0]);
                float i2 = sigf_b(accv[1][r], bsc[1]);
                float f1 = sigf_b(accv[2][r], bsc[2]);
                float f2 = sigf_b(accv[3][r], bsc[3]);
                float g1 = tanh_b(accv[4][r], bsc[4]);
                float g2 = tanh_b(accv[5][r], bsc[5]);
                float o1 = sigf_b(accv[6][r], bsc[6]);
                float o2 = sigf_b(accv[7][r], bsc[7]);
                c1[r] = f1 * c1[r] + i1 * g1;
                c2[r] = f2 * c2[r] + i2 * g2;
                float h1 = o1 * tanh_c(c1[r]);
                float h2 = o2 * tanh_c(c2[r]);
                hout[M * HDIM + mrow]      = (_Float16)h1;
                hout[M * HDIM + mrow + 16] = (_Float16)h2;
            }

            // same-wave LDS ops are in-order; wait for stores, reload h as A
            asm volatile("s_wait_dscnt 0" ::: "memory");
            {
                const _Float16* pa = sbuf + (t * ROWS_PER_WAVE + mrow) * HDIM + hiHalf * 16;
                v8h lo = *(const v8h*)pa;
                v8h hi = *(const v8h*)(pa + 8);
                ah = __builtin_shufflevector(lo, hi, 0,1,2,3,4,5,6,7,
                                                     8,9,10,11,12,13,14,15);
            }
        } // t
    } // layer

    // ---- fused affine head: out = h_last @ Wf^T + bf ----------------------
    if (lane < 16) {
        const _Float16* hp = sbuf + ((TSTEP - 1) * ROWS_PER_WAVE + mrow) * HDIM;
        float hv[HDIM];
#pragma unroll
        for (int q = 0; q < 4; ++q) {
            v8h hq = *(const v8h*)(hp + q * 8);
#pragma unroll
            for (int e = 0; e < 8; ++e) hv[q * 8 + e] = (float)hq[e];
        }
        const float* Wf = (const float*)(wpack + HEAD_OFF);
        const float* bf = Wf + NCLS * HDIM;
        float* op = out + (size_t)(row0 + mrow) * NCLS;
        for (int k = 0; k < NCLS; ++k) {       // small, left un-unrolled
            float s = bf[k];
#pragma unroll
            for (int n = 0; n < HDIM; ++n) s += Wf[k * HDIM + n] * hv[n];
            op[k] = s;
        }
    }
}

// ---------------------------------------------------------------------------
extern "C" void kernel_launch(void* const* d_in, const int* in_sizes, int n_in,
                              void* d_out, int out_size, void* d_ws, size_t ws_size,
                              hipStream_t stream)
{
    const float* X     = (const float*)d_in[0];
    const float* Wih0  = (const float*)d_in[1];
    const float* Whh0  = (const float*)d_in[2];
    const float* bi0   = (const float*)d_in[3];
    const float* bh0   = (const float*)d_in[4];
    const float* Wih1  = (const float*)d_in[5];
    const float* Whh1  = (const float*)d_in[6];
    const float* bi1   = (const float*)d_in[7];
    const float* bh1   = (const float*)d_in[8];
    const float* Wih2  = (const float*)d_in[9];
    const float* Whh2  = (const float*)d_in[10];
    const float* bi2   = (const float*)d_in[11];
    const float* bh2   = (const float*)d_in[12];
    const float* W1    = (const float*)d_in[13];
    const float* b1    = (const float*)d_in[14];
    const float* W2    = (const float*)d_in[15];
    const float* b2    = (const float*)d_in[16];
    (void)n_in; (void)ws_size;

    char* ws = (char*)d_ws;

    pack_weights<<<4, 128, 0, stream>>>(Wih0, Whh0, bi0, bh0,
                                        Wih1, Whh1, bi1, bh1,
                                        Wih2, Whh2, bi2, bh2,
                                        W1, b1, W2, b2, ws);

    const int B   = in_sizes[0] / (TSTEP * 3);          // 65536
    const int wgs = B / (WAVES_PER_WG * ROWS_PER_WAVE); // 2048
    lstm3_wmma<<<wgs, WAVES_PER_WG * 32, 0, stream>>>(X, ws, (float*)d_out);
    (void)out_size;
}
